// BEVFormer_68805376082092
// MI455X (gfx1250) — matrix operand order repaired and checked by reading, not compiled, not run on hardware
//
#include <hip/hip_runtime.h>
#include <cstddef>

typedef __attribute__((ext_vector_type(16))) _Float16 v16h;
typedef __attribute__((ext_vector_type(4)))  _Float16 v4h;
typedef __attribute__((ext_vector_type(8)))  float    v8f;
typedef __attribute__((ext_vector_type(4)))  unsigned int v4u;
typedef __attribute__((ext_vector_type(4)))  int      v4i;
typedef __attribute__((ext_vector_type(8)))  int      v8i;

#define Qn   6400
#define Cn   256
#define NHn  8
#define HDn  32
#define Vn   6
#define Sn   7979
#define Zn   4
#define Ln   4
#define Pn   2
#define FFn  512
#define BEVn 80
#define WORLD_EXT 40.96f

// ---------------- conversion kernels ----------------
// vectorized: n must be a multiple of 4 (all our tensors are)
__global__ void __launch_bounds__(256) k_f32_to_f16(const float4* __restrict__ x,
                                                    v4h* __restrict__ y, int n4) {
  int i = blockIdx.x * 256 + threadIdx.x;
  if (i < n4) {
    float4 v = x[i];
    v4h h;
    h[0] = (_Float16)v.x; h[1] = (_Float16)v.y;
    h[2] = (_Float16)v.z; h[3] = (_Float16)v.w;
    y[i] = h;
  }
}

// x is (K,N) row-major f32; y becomes (N,K) row-major f16 (i.e. col-major weights)
__global__ void __launch_bounds__(256) k_conv_transpose(const float* __restrict__ x,
                                                        _Float16* __restrict__ y,
                                                        int K, int N) {
  int i = blockIdx.x * 256 + threadIdx.x;
  if (i < K * N) {
    int k = i / N, n = i - k * N;
    y[(size_t)n * K + k] = (_Float16)x[i];
  }
}

// ---------------- WMMA GEMM with TDM-staged weight panel ----------------
// C[M,N] = A[M,K] (f16 row-major) x W (f16, stored transposed Bt[N,K]) + bias, opt relu.
// Block: 256 threads = 8 waves; block tile 128(M) x 64(N); wave tile 16 x 64.
// The 64xK B panel is DMA'd into LDS once per block by the Tensor Data Mover.
__global__ void __launch_bounds__(256) k_wmma_gemm(
    const _Float16* __restrict__ A, const _Float16* __restrict__ Bt,
    const float* __restrict__ bias, float* __restrict__ Co,
    int M, int N, int K, int relu)
{
  __shared__ _Float16 lB[64 * 512];            // up to 64 rows x K<=512 halves = 64KB
  const int tid  = threadIdx.x;
  const int lane = tid & 31, wave = tid >> 5;
  const int m0   = blockIdx.y * 128 + wave * 16;
  const int nb0  = blockIdx.x * 64;
  const int half = lane >> 4;                  // lane half selects K sub-block
  const int idx  = lane & 15;                  // A row / B col within tile

  // --- TDM: one wave issues the descriptor; OOB rows (>= N-nb0) are zero-filled ---
  if (wave == 0) {
    unsigned long long gaddr = (unsigned long long)(const void*)(Bt + (size_t)nb0 * K);
    unsigned ldsb = (unsigned)(unsigned long long)(const void*)&lB[0];
    v4u g0;
    g0[0] = 1u;                                            // count = 1 valid descriptor
    g0[1] = ldsb;                                          // lds_addr (bytes)
    g0[2] = (unsigned)(gaddr & 0xFFFFFFFFull);             // global_addr[31:0]
    g0[3] = (unsigned)((gaddr >> 32) & 0x1FFFFFFull)       // global_addr[56:32]
            | 0x80000000u;                                 // type = 2 ("image")
    v8i g1;
    g1[0] = (int)(1u << 16);                               // data_size=1 -> 2 bytes
    g1[1] = (int)((unsigned)K << 16);                      // tensor_dim0[15:0]
    g1[2] = (int)(((unsigned)(N - nb0) & 0xFFFFu) << 16);  // tensor_dim1[15:0]
    g1[3] = (int)((unsigned)K << 16);                      // tile_dim0 = K
    g1[4] = 64;                                            // tile_dim1 = 64 rows
    g1[5] = K;                                             // tensor_dim0_stride
    g1[6] = 0;
    g1[7] = 0;
    v4i gz = {0, 0, 0, 0};
    v8i gz8 = {0, 0, 0, 0, 0, 0, 0, 0};
    __builtin_amdgcn_tensor_load_to_lds(g0, g1, gz, gz, gz8, 0);
    __builtin_amdgcn_s_wait_tensorcnt(0);
  }
  __syncthreads();

  // A row: clamp instead of branch — OOB A rows only feed OOB D rows (masked at store)
  int arow = m0 + idx;
  if (arow >= M) arow = M - 1;
  const _Float16* arp = A + (size_t)arow * K;

  v8f acc[4] = {};
  for (int kb = 0; kb < K; kb += 32) {
    // A fragment: lanes 0-15 hold K=kb+{0..15}; lanes 16-31 K=kb+{16..31}
    const int ka = kb + half * 16;
    __builtin_prefetch(arp + ka + 64, 0, 3);   // global_prefetch_b8 for next chunk
    v16h a;
#pragma unroll
    for (int e = 0; e < 16; ++e) a[e] = arp[ka + e];
    // 4 N-tiles reuse the same A fragment
#pragma unroll
    for (int t = 0; t < 4; ++t) {
      const _Float16* lp = &lB[(size_t)(t * 16 + idx) * K + kb + half * 16];
      v16h b;
#pragma unroll
      for (int e = 0; e < 16; ++e) b[e] = lp[e];
      acc[t] = __builtin_amdgcn_wmma_f32_16x16x32_f16(false, a, false, b,
                                                      (short)0, acc[t], false, false);
    }
  }
  // D layout: VGPR r -> row m0 + r + 8*half, col n0 + idx
#pragma unroll
  for (int t = 0; t < 4; ++t) {
    int col = nb0 + t * 16 + idx;
    if (col < N) {
#pragma unroll
      for (int r = 0; r < 8; ++r) {
        int row = m0 + r + half * 8;
        if (row < M) {
          float v = acc[t][r];
          if (bias) v += bias[col];
          if (relu) v = fmaxf(v, 0.f);
          Co[(size_t)row * N + col] = v;
        }
      }
    }
  }
}

// ---------------- residual + LayerNorm ----------------
__global__ void __launch_bounds__(256) k_add_ln(const float* __restrict__ x,
                                                const float* __restrict__ r,
                                                const float* __restrict__ g,
                                                const float* __restrict__ bb,
                                                float* __restrict__ y) {
  int row = blockIdx.x, t = threadIdx.x;
  float v = x[(size_t)row * Cn + t] + r[(size_t)row * Cn + t];
  __shared__ float s1[256], s2[256];
  s1[t] = v; s2[t] = v * v;
  __syncthreads();
  for (int o = 128; o > 0; o >>= 1) {
    if (t < o) { s1[t] += s1[t + o]; s2[t] += s2[t + o]; }
    __syncthreads();
  }
  float mean = s1[0] * (1.f / Cn);
  float var  = s2[0] * (1.f / Cn) - mean * mean;
  float inv  = rsqrtf(var + 1e-5f);
  y[(size_t)row * Cn + t] = (v - mean) * inv * g[t] + bb[t];
}

// ---------------- bilinear sample (OOB corners contribute zero) ----------------
__device__ __forceinline__ float bilin(const float* __restrict__ base, int H, int W,
                                       int ch, float lx, float ly) {
  float x = lx * W - 0.5f, y = ly * H - 0.5f;
  float xf = floorf(x), yf = floorf(y);
  float fx = x - xf, fy = y - yf;
  int x0 = (int)xf, y0 = (int)yf;
  float r = 0.f;
#pragma unroll
  for (int dy = 0; dy < 2; ++dy) {
    int yi = y0 + dy;
    if (yi < 0 || yi >= H) continue;
    float wy = dy ? fy : 1.f - fy;
#pragma unroll
    for (int dx = 0; dx < 2; ++dx) {
      int xi = x0 + dx;
      if (xi < 0 || xi >= W) continue;
      float wx = dx ? fx : 1.f - fx;
      r += wy * wx * base[((size_t)(yi * W + xi)) * Cn + ch];
    }
  }
  return r;
}

// ---------------- temporal deformable attention ----------------
__global__ void __launch_bounds__(256) k_temporal(
    const float* __restrict__ vcur, const float* __restrict__ vhist,
    const float* __restrict__ offt, const float* __restrict__ wlog,
    const float* __restrict__ Tm, float* __restrict__ outt)
{
  int q = blockIdx.x;
  int d = threadIdx.x & 31, h = threadIdx.x >> 5;
  int qx = q % BEVn, qy = q / BEVn;
  float rx = (qx + 0.5f) / BEVn, ry = (qy + 0.5f) / BEVn;
  float wx = (rx - 0.5f) * WORLD_EXT, wy = (ry - 0.5f) * WORLD_EXT;
  float hz  = Tm[6] * wx + Tm[7] * wy + Tm[8];
  float rhx = (Tm[0] * wx + Tm[1] * wy + Tm[2]) / hz / WORLD_EXT + 0.5f;
  float rhy = (Tm[3] * wx + Tm[4] * wy + Tm[5]) / hz / WORLD_EXT + 0.5f;

  const float* wl = wlog + (size_t)q * (NHn * 4) + h * 4;
  float l0 = wl[0], l1 = wl[1], l2 = wl[2], l3 = wl[3];
  float mx = fmaxf(fmaxf(l0, l1), fmaxf(l2, l3));
  float e0 = __expf(l0 - mx), e1 = __expf(l1 - mx),
        e2 = __expf(l2 - mx), e3 = __expf(l3 - mx);
  float inv = 1.f / (e0 + e1 + e2 + e3);
  float w[4] = {e0 * inv, e1 * inv, e2 * inv, e3 * inv};

  const float* ob = offt + (size_t)q * 64 + h * 8;   // (2 branches, P, 2)
  int ch = h * HDn + d;
  float acc = 0.f;
#pragma unroll
  for (int br = 0; br < 2; ++br) {
    const float* val = br ? vhist : vcur;
    float bx = br ? rhx : rx, by = br ? rhy : ry;
#pragma unroll
    for (int p = 0; p < Pn; ++p) {
      float ox = ob[br * 4 + p * 2 + 0] * (1.f / BEVn);
      float oy = ob[br * 4 + p * 2 + 1] * (1.f / BEVn);
      acc += w[br * 2 + p] * bilin(val, BEVn, BEVn, ch, bx + ox, by + oy);
    }
  }
  outt[(size_t)q * Cn + ch] = acc;
}

// ---------------- spatial (multi-view multi-scale) deformable attention ----------------
__global__ void __launch_bounds__(256) k_spatial(
    const float* __restrict__ val,  const float* __restrict__ offs,
    const float* __restrict__ wlog, const float* __restrict__ Pm,
    const float* __restrict__ zr,   float* __restrict__ outp)
{
  const int LH[4] = {60, 30, 15, 8};
  const int LW[4] = {100, 50, 25, 13};
  const int LS[4] = {0, 6000, 7500, 7875};
  int q = blockIdx.x;
  int d = threadIdx.x & 31, h = threadIdx.x >> 5;   // one head per wave32
  int qx = q % BEVn, qy = q / BEVn;
  float wx = ((qx + 0.5f) / BEVn - 0.5f) * WORLD_EXT;
  float wy = ((qy + 0.5f) / BEVn - 0.5f) * WORLD_EXT;

  // softmax over 32 logits per (q,h) via wave32 shuffles; cache in LDS
  const float* wl = wlog + (size_t)q * 256 + h * 32;
  float lg = wl[d];
  float mx = lg;
#pragma unroll
  for (int o = 16; o > 0; o >>= 1) mx = fmaxf(mx, __shfl_xor(mx, o));
  float ex = __expf(lg - mx);
  float s = ex;
#pragma unroll
  for (int o = 16; o > 0; o >>= 1) s += __shfl_xor(s, o);
  __shared__ float sw[256];
  sw[h * 32 + d] = ex / s;
  __syncthreads();

  const float* ob = offs + (size_t)q * 512 + h * 64; // (Z, L, P, 2)
  int ch = h * HDn + d;
  float acc = 0.f, cnt = 0.f;
  for (int v = 0; v < Vn; ++v) {
    const float* Pv = Pm + v * 12;
    for (int z = 0; z < Zn; ++z) {
      float zz = zr[z];
      float u  = Pv[0] * wx + Pv[1] * wy + Pv[2]  * zz + Pv[3];
      float vv = Pv[4] * wx + Pv[5] * wy + Pv[6]  * zz + Pv[7];
      float dd = Pv[8] * wx + Pv[9] * wy + Pv[10] * zz + Pv[11];
      float dm = fmaxf(dd, 1e-5f);
      float un = u  / dm * (1.f / 800.f);
      float vn = vv / dm * (1.f / 480.f);
      bool valid = (dd > 1e-5f) && (un >= 0.f) && (un <= 1.f) &&
                   (vn >= 0.f) && (vn <= 1.f);
      if (!valid) continue;                       // uniform across block
      cnt += 1.f;
#pragma unroll
      for (int l = 0; l < Ln; ++l) {
        const float* base = val + ((size_t)v * Sn + LS[l]) * Cn;
        float invW = 1.f / LW[l], invH = 1.f / LH[l];
#pragma unroll
        for (int p = 0; p < Pn; ++p) {
          float ox = ob[z * 16 + l * 4 + p * 2 + 0] * invW;
          float oy = ob[z * 16 + l * 4 + p * 2 + 1] * invH;
          float w  = sw[h * 32 + z * 8 + l * 2 + p];
          acc += w * bilin(base, LH[l], LW[l], ch, un + ox, vn + oy);
        }
      }
    }
  }
  cnt = fmaxf(cnt, 1.f);
  outp[(size_t)q * Cn + ch] = acc / cnt;
}

// ---------------- host orchestration ----------------
extern "C" void kernel_launch(void* const* d_in, const int* in_sizes, int n_in,
                              void* d_out, int out_size, void* d_ws, size_t ws_size,
                              hipStream_t stream) {
  (void)in_sizes; (void)n_in; (void)out_size; (void)ws_size;
  const float* q      = (const float*)d_in[0];
  const float* hist   = (const float*)d_in[1];
  const float* fmaps  = (const float*)d_in[2];
  const float* Tm     = (const float*)d_in[3];
  const float* zr     = (const float*)d_in[4];
  const float* Pm     = (const float*)d_in[5];
  const float* Wv_t   = (const float*)d_in[6];
  const float* Woff_t = (const float*)d_in[7];
  const float* boff_t = (const float*)d_in[8];
  const float* Ww_t   = (const float*)d_in[9];
  const float* bw_t   = (const float*)d_in[10];
  const float* Wo_t   = (const float*)d_in[11];
  const float* bo_t   = (const float*)d_in[12];
  const float* ln1g   = (const float*)d_in[13];
  const float* ln1b   = (const float*)d_in[14];
  const float* Wv_s   = (const float*)d_in[15];
  const float* Woff_s = (const float*)d_in[16];
  const float* boff_s = (const float*)d_in[17];
  const float* Ww_s   = (const float*)d_in[18];
  const float* bw_s   = (const float*)d_in[19];
  const float* Wo_s   = (const float*)d_in[20];
  const float* bo_s   = (const float*)d_in[21];
  const float* ln2g   = (const float*)d_in[22];
  const float* ln2b   = (const float*)d_in[23];
  const float* W1     = (const float*)d_in[24];
  const float* b1     = (const float*)d_in[25];
  const float* W2     = (const float*)d_in[26];
  const float* b2     = (const float*)d_in[27];
  const float* ln3g   = (const float*)d_in[28];
  const float* ln3b   = (const float*)d_in[29];

  size_t arena = 0;
  auto alloc = [&](size_t bytes) -> void* {
    void* p = (char*)d_ws + arena;
    arena += (bytes + 255) & ~(size_t)255;
    return p;
  };
  const size_t QC  = (size_t)Qn * Cn;
  const size_t FM  = (size_t)Vn * Sn * Cn;
  const int    Mfm = Vn * Sn;

  _Float16* q16     = (_Float16*)alloc(QC * 2);
  _Float16* hist16  = (_Float16*)alloc(QC * 2);
  _Float16* fm16    = (_Float16*)alloc(FM * 2);
  _Float16* wvt16   = (_Float16*)alloc((size_t)Cn * Cn  * 2);
  _Float16* wofft16 = (_Float16*)alloc((size_t)Cn * 64  * 2);
  _Float16* wwt16   = (_Float16*)alloc((size_t)Cn * 32  * 2);
  _Float16* wot16   = (_Float16*)alloc((size_t)Cn * Cn  * 2);
  _Float16* wvs16   = (_Float16*)alloc((size_t)Cn * Cn  * 2);
  _Float16* woffs16 = (_Float16*)alloc((size_t)Cn * 512 * 2);
  _Float16* wws16   = (_Float16*)alloc((size_t)Cn * 256 * 2);
  _Float16* wos16   = (_Float16*)alloc((size_t)Cn * Cn  * 2);
  _Float16* w116    = (_Float16*)alloc((size_t)Cn * FFn * 2);
  _Float16* w216    = (_Float16*)alloc((size_t)FFn * Cn * 2);
  float* vcur   = (float*)alloc(QC * 4);
  float* vhist  = (float*)alloc(QC * 4);
  float* offt   = (float*)alloc((size_t)Qn * 64 * 4);
  float* wtl    = (float*)alloc((size_t)Qn * 32 * 4);
  float* tout   = (float*)alloc(QC * 4);
  _Float16* tout16 = (_Float16*)alloc(QC * 2);
  float* out1   = (float*)alloc(QC * 4);
  float* out2   = (float*)alloc(QC * 4);
  _Float16* out216 = (_Float16*)alloc(QC * 2);
  float* val    = (float*)alloc(FM * 4);
  float* offsb  = (float*)alloc((size_t)Qn * 512 * 4);
  float* wsl    = (float*)alloc((size_t)Qn * 256 * 4);
  float* sacc   = (float*)alloc(QC * 4);
  _Float16* sacc16 = (_Float16*)alloc(QC * 2);
  float* out3   = (float*)alloc(QC * 4);
  float* out4   = (float*)alloc(QC * 4);
  _Float16* out416 = (_Float16*)alloc(QC * 2);
  float* ffh    = (float*)alloc((size_t)Qn * FFn * 4);
  _Float16* ffh16  = (_Float16*)alloc((size_t)Qn * FFn * 2);
  float* out5   = (float*)alloc(QC * 4);

  auto conv = [&](const float* x, _Float16* y, size_t n) {
    size_t n4 = n / 4;
    k_f32_to_f16<<<dim3((unsigned)((n4 + 255) / 256)), dim3(256), 0, stream>>>(
        (const float4*)x, (v4h*)y, (int)n4);
  };
  auto convT = [&](const float* x, _Float16* y, int K, int N) {
    k_conv_transpose<<<dim3((unsigned)(((size_t)K * N + 255) / 256)), dim3(256), 0, stream>>>(x, y, K, N);
  };
  auto gemm = [&](const _Float16* A, const _Float16* Bt, const float* bias, float* Cc,
                  int M, int N, int K, int relu) {
    dim3 grid((N + 63) / 64, (M + 127) / 128);
    k_wmma_gemm<<<grid, dim3(256), 0, stream>>>(A, Bt, bias, Cc, M, N, K, relu);
  };

  // precision conversion
  conv(q, q16, QC);
  conv(hist, hist16, QC);
  conv(fmaps, fm16, FM);
  convT(Wv_t, wvt16, Cn, Cn);
  convT(Woff_t, wofft16, Cn, 64);
  convT(Ww_t, wwt16, Cn, 32);
  convT(Wo_t, wot16, Cn, Cn);
  convT(Wv_s, wvs16, Cn, Cn);
  convT(Woff_s, woffs16, Cn, 512);
  convT(Ww_s, wws16, Cn, 256);
  convT(Wo_s, wos16, Cn, Cn);
  convT(W1, w116, Cn, FFn);
  convT(W2, w216, FFn, Cn);

  // temporal self-attention
  gemm(q16,    wvt16,   nullptr, vcur,  Qn, Cn, Cn, 0);
  gemm(hist16, wvt16,   nullptr, vhist, Qn, Cn, Cn, 0);
  gemm(q16,    wofft16, boff_t,  offt,  Qn, 64, Cn, 0);
  gemm(q16,    wwt16,   bw_t,    wtl,   Qn, 32, Cn, 0);
  k_temporal<<<dim3(Qn), dim3(256), 0, stream>>>(vcur, vhist, offt, wtl, Tm, tout);
  conv(tout, tout16, QC);
  gemm(tout16, wot16, bo_t, out1, Qn, Cn, Cn, 0);
  k_add_ln<<<dim3(Qn), dim3(256), 0, stream>>>(out1, q, ln1g, ln1b, out2);
  conv(out2, out216, QC);

  // spatial cross-attention
  gemm(fm16,   wvs16,   nullptr, val,   Mfm, Cn,  Cn, 0);
  gemm(out216, woffs16, boff_s,  offsb, Qn,  512, Cn, 0);
  gemm(out216, wws16,   bw_s,    wsl,   Qn,  256, Cn, 0);
  k_spatial<<<dim3(Qn), dim3(256), 0, stream>>>(val, offsb, wsl, Pm, zr, sacc);
  conv(sacc, sacc16, QC);
  gemm(sacc16, wos16, bo_s, out3, Qn, Cn, Cn, 0);
  k_add_ln<<<dim3(Qn), dim3(256), 0, stream>>>(out3, out2, ln2g, ln2b, out4);
  conv(out4, out416, QC);

  // FFN
  gemm(out416, w116, b1, ffh, Qn, FFn, Cn, 1);
  conv(ffh, ffh16, (size_t)Qn * FFn);
  gemm(ffh16, w216, b2, out5, Qn, Cn, FFn, 0);
  k_add_ln<<<dim3(Qn), dim3(256), 0, stream>>>(out5, out4, ln3g, ln3b, (float*)d_out);
}
